// ExprGraphNet_27298812133379
// MI455X (gfx1250) — matrix operand order repaired and checked by reading, not compiled, not run on hardware
//
#include <hip/hip_runtime.h>

// ---------------- types ----------------
typedef __attribute__((ext_vector_type(16))) __bf16 v16bf;
typedef __attribute__((ext_vector_type(8)))  __bf16 v8bf;
typedef __attribute__((ext_vector_type(8)))  float  v8f;

#define NN   131072
#define EE   524288
#define GG   1024
#define HID  128

__device__ __forceinline__ unsigned short f2bf(float f) {
  unsigned int u = __float_as_uint(f);
  unsigned int r = 0x7FFFu + ((u >> 16) & 1u);   // round-to-nearest-even
  return (unsigned short)((u + r) >> 16);
}

// order-preserving float -> uint map for atomicMax (memset(0) == -inf identity)
__device__ __forceinline__ unsigned int encMax(float f) {
  unsigned int u = __float_as_uint(f);
  return (u & 0x80000000u) ? ~u : (u | 0x80000000u);
}
__device__ __forceinline__ float decMax(unsigned int u) {
  return (u & 0x80000000u) ? __uint_as_float(u & 0x7FFFFFFFu)
                           : __uint_as_float(~u);
}

// ---------------- WMMA GEMM: out[M,128] = A[M,K] @ W[K,128] + bias ----------
// Kp: K padded to multiple of 32 (<=128). 256 threads = 8 waves, each wave
// computes a 16x16 tile of the 16x128 row-tile. Weights staged to LDS bf16
// column-major; A tile staged row-major.
__global__ __launch_bounds__(256) void gemm_bias_wmma(
    const float* __restrict__ A, int lda, int K, int Kp,
    const float* __restrict__ W, const float* __restrict__ bias,
    float* __restrict__ out, int M)
{
  __shared__ unsigned short sW[128 * 128];  // sW[n*128 + k] (col-major)
  __shared__ unsigned short sA[16 * 128];   // sA[m*128 + k] (row-major)

  const int tid  = threadIdx.x;
  const int wave = tid >> 5;
  const int lane = tid & 31;
  const int half = lane >> 4;
  const int l16  = lane & 15;

  // stage weights once per block (L2-resident across blocks)
  for (int i = tid; i < Kp * 128; i += 256) {
    int k = i >> 7, n = i & 127;
    float v = 0.0f;
    if (k < K) v = W[k * 128 + n];
    sW[n * 128 + k] = f2bf(v);
  }
  __syncthreads();

  const int numTiles = M >> 4;
  const int colBase  = wave * 16;
  const float bcol   = bias[colBase + l16];

  for (int tile = blockIdx.x; tile < numTiles; tile += gridDim.x) {
    for (int i = tid; i < 16 * Kp; i += 256) {
      int m = i / Kp, k = i - m * Kp;
      float v = 0.0f;
      if (k < K) v = A[(tile * 16 + m) * lda + k];
      sA[m * 128 + k] = f2bf(v);
    }
    __syncthreads();

    v8f acc = {};
    for (int k0 = 0; k0 < Kp; k0 += 32) {
      // A fragment: 16-bit 16x32 layout (two contiguous 8-half runs per lane)
      v8bf alo = *(const v8bf*)(&sA[l16 * 128 + k0 + half * 8]);
      v8bf ahi = *(const v8bf*)(&sA[l16 * 128 + k0 + 16 + half * 8]);
      v16bf a;
      #pragma unroll
      for (int i = 0; i < 8; ++i) { a[i] = alo[i]; a[i + 8] = ahi[i]; }
      // B fragment: lane = column, same k order as A (two 8-runs: half*8, 16+half*8)
      v8bf blo = *(const v8bf*)(&sW[(colBase + l16) * 128 + k0 + half * 8]);
      v8bf bhi = *(const v8bf*)(&sW[(colBase + l16) * 128 + k0 + 16 + half * 8]);
      v16bf b;
      #pragma unroll
      for (int i = 0; i < 8; ++i) { b[i] = blo[i]; b[i + 8] = bhi[i]; }
      acc = __builtin_amdgcn_wmma_f32_16x16x32_bf16(
          false, a, false, b, (short)0, acc, false, false);
    }
    // C/D layout: element v -> row = v + half*8, col = colBase + l16
    #pragma unroll
    for (int v = 0; v < 8; ++v) {
      int row = tile * 16 + v + half * 8;
      out[row * 128 + colBase + l16] = acc[v] + bcol;
    }
    __syncthreads();
  }
}

// ------------- edge kernels (1 wave32 per edge; lane = channel) -------------
__global__ __launch_bounds__(256) void edge_score_max(
    const float* __restrict__ hl, const float* __restrict__ hr,
    const float* __restrict__ ea, const float* __restrict__ We,
    const float* __restrict__ att, const int* __restrict__ src,
    const int* __restrict__ dst, float* __restrict__ score,
    unsigned int* __restrict__ mxe, int E)
{
  int gw   = (blockIdx.x * 256 + threadIdx.x) >> 5;
  int lane = threadIdx.x & 31;
  if (gw >= E) return;
  int s = src[gw], d = dst[gw];
  float eav = ea[gw];
  #pragma unroll
  for (int h = 0; h < 4; ++h) {
    int c = h * 32 + lane;
    float m = hl[s * 128 + c] + hr[d * 128 + c] + eav * We[c];
    m = (m > 0.0f) ? m : 0.2f * m;              // leaky_relu 0.2
    float p = m * att[c];
    for (int off = 16; off > 0; off >>= 1) p += __shfl_xor(p, off, 32);
    if (lane == 0) {
      score[gw * 4 + h] = p;
      atomicMax(&mxe[d * 4 + h], encMax(p));
    }
  }
}

__global__ __launch_bounds__(256) void edge_exp(
    const float* __restrict__ score, const unsigned int* __restrict__ mxe,
    const int* __restrict__ dst, float* __restrict__ exsc,
    float* __restrict__ den, int EH)
{
  int i = blockIdx.x * 256 + threadIdx.x;
  if (i >= EH) return;
  int e = i >> 2, h = i & 3;
  int d = dst[e];
  float ex = __expf(score[i] - decMax(mxe[d * 4 + h]));
  exsc[i] = ex;
  atomicAdd(&den[d * 4 + h], ex);
}

__global__ __launch_bounds__(256) void edge_aggr(
    const float* __restrict__ hl, const float* __restrict__ exsc,
    const float* __restrict__ den, const int* __restrict__ src,
    const int* __restrict__ dst, float* __restrict__ aggr, int E)
{
  int gw   = (blockIdx.x * 256 + threadIdx.x) >> 5;
  int lane = threadIdx.x & 31;
  if (gw >= E) return;
  int s = src[gw], d = dst[gw];
  #pragma unroll
  for (int h = 0; h < 4; ++h) {
    float alpha = exsc[gw * 4 + h] / den[d * 4 + h];
    int c = h * 32 + lane;
    atomicAdd(&aggr[d * 128 + c], alpha * hl[s * 128 + c]);
  }
}

// ------------- residual + layernorm (1 wave per node), in place -------------
__global__ __launch_bounds__(256) void residual_ln(
    float* __restrict__ h, const float* __restrict__ aggr,
    const float* __restrict__ cb, const float* __restrict__ g,
    const float* __restrict__ b, int Nn)
{
  int gw   = (blockIdx.x * 256 + threadIdx.x) >> 5;
  int lane = threadIdx.x & 31;
  if (gw >= Nn) return;
  float v[4]; float s = 0.0f;
  #pragma unroll
  for (int j = 0; j < 4; ++j) {
    int c = j * 32 + lane;
    v[j] = h[gw * 128 + c] + aggr[gw * 128 + c] + cb[c];
    s += v[j];
  }
  for (int off = 16; off > 0; off >>= 1) s += __shfl_xor(s, off, 32);
  float mu = s * (1.0f / 128.0f);
  float var = 0.0f;
  #pragma unroll
  for (int j = 0; j < 4; ++j) { float d = v[j] - mu; var += d * d; }
  for (int off = 16; off > 0; off >>= 1) var += __shfl_xor(var, off, 32);
  float rs = rsqrtf(var * (1.0f / 128.0f) + 1e-5f);
  #pragma unroll
  for (int j = 0; j < 4; ++j) {
    int c = j * 32 + lane;
    h[gw * 128 + c] = (v[j] - mu) * rs * g[c] + b[c];
  }
}

// ------------- global mean pool + value head --------------------------------
__global__ __launch_bounds__(256) void pool_sum(
    const float* __restrict__ h, const int* __restrict__ batch,
    float* __restrict__ sums, float* __restrict__ cnt, int Nn)
{
  int gw   = (blockIdx.x * 256 + threadIdx.x) >> 5;
  int lane = threadIdx.x & 31;
  if (gw >= Nn) return;
  int g = batch[gw];
  #pragma unroll
  for (int j = 0; j < 4; ++j) {
    int c = j * 32 + lane;
    atomicAdd(&sums[g * 128 + c], h[gw * 128 + c]);
  }
  if (lane == 0) atomicAdd(&cnt[g], 1.0f);
}

__global__ __launch_bounds__(256) void value_head(
    const float* __restrict__ sums, const float* __restrict__ cnt,
    const float* __restrict__ W1, const float* __restrict__ b1,
    const float* __restrict__ W2, const float* __restrict__ b2,
    float* __restrict__ out, int G)
{
  int gw   = (blockIdx.x * 256 + threadIdx.x) >> 5;
  int lane = threadIdx.x & 31;
  if (gw >= G) return;
  float inv = 1.0f / fmaxf(cnt[gw], 1.0f);
  float v = 0.0f;
  #pragma unroll
  for (int uu = 0; uu < 2; ++uu) {
    int u = lane + uu * 32;
    float acc = b1[u];
    for (int k = 0; k < 128; ++k)
      acc += (sums[gw * 128 + k] * inv) * W1[k * 64 + u];
    acc = fmaxf(acc, 0.0f);
    v += acc * W2[u];
  }
  for (int off = 16; off > 0; off >>= 1) v += __shfl_xor(v, off, 32);
  if (lane == 0) out[gw] = v + b2[0];
}

// ---------------- host launcher ----------------
extern "C" void kernel_launch(void* const* d_in, const int* in_sizes, int n_in,
                              void* d_out, int out_size, void* d_ws, size_t ws_size,
                              hipStream_t stream) {
  (void)in_sizes; (void)n_in; (void)out_size; (void)ws_size;
  const float* x    = (const float*)d_in[0];
  const int*   ei   = (const int*)  d_in[1];
  const float* ea   = (const float*)d_in[2];
  const int*   bat  = (const int*)  d_in[3];
  const float* W_in = (const float*)d_in[4];
  const float* b_in = (const float*)d_in[5];
  const float* Wl   = (const float*)d_in[6];
  const float* bl   = (const float*)d_in[7];
  const float* Wr   = (const float*)d_in[8];
  const float* br   = (const float*)d_in[9];
  const float* We   = (const float*)d_in[10];
  const float* att  = (const float*)d_in[11];
  const float* cb   = (const float*)d_in[12];
  const float* lng  = (const float*)d_in[13];
  const float* lnb  = (const float*)d_in[14];
  const float* W1   = (const float*)d_in[15];
  const float* b1   = (const float*)d_in[16];
  const float* W2   = (const float*)d_in[17];
  const float* b2   = (const float*)d_in[18];
  float* out = (float*)d_out;

  const int* src = ei;
  const int* dst = ei + EE;

  char* ws = (char*)d_ws;
  size_t off = 0;
  auto carve = [&](size_t bytes) {
    void* p = ws + off; off += (bytes + 255) & ~size_t(255); return p;
  };
  float*        h    = (float*)       carve((size_t)NN * 128 * 4);
  float*        hl   = (float*)       carve((size_t)NN * 128 * 4);
  float*        hr   = (float*)       carve((size_t)NN * 128 * 4);
  float*        aggr = (float*)       carve((size_t)NN * 128 * 4);
  float*        sc   = (float*)       carve((size_t)EE * 4 * 4);
  float*        exs  = (float*)       carve((size_t)EE * 4 * 4);
  unsigned int* mxe  = (unsigned int*)carve((size_t)NN * 4 * 4);
  float*        den  = (float*)       carve((size_t)NN * 4 * 4);
  float*        sums = (float*)       carve((size_t)GG * 128 * 4);
  float*        cnt  = (float*)       carve((size_t)GG * 4);

  // input projection: [N,44] @ [44,128]  (K padded 44 -> 64)
  gemm_bias_wmma<<<512, 256, 0, stream>>>(x, 44, 44, 64, W_in, b_in, h, NN);

  for (int l = 0; l < 4; ++l) {
    gemm_bias_wmma<<<512, 256, 0, stream>>>(h, 128, 128, 128,
                                            Wl + (size_t)l * 128 * 128,
                                            bl + l * 128, hl, NN);
    gemm_bias_wmma<<<512, 256, 0, stream>>>(h, 128, 128, 128,
                                            Wr + (size_t)l * 128 * 128,
                                            br + l * 128, hr, NN);
    hipMemsetAsync(mxe,  0, (size_t)NN * 4 * 4,   stream);  // encMax identity
    hipMemsetAsync(den,  0, (size_t)NN * 4 * 4,   stream);
    hipMemsetAsync(aggr, 0, (size_t)NN * 128 * 4, stream);
    edge_score_max<<<EE * 32 / 256, 256, 0, stream>>>(
        hl, hr, ea, We + l * 128, att + l * 128, src, dst, sc, mxe, EE);
    edge_exp<<<EE * 4 / 256, 256, 0, stream>>>(sc, mxe, dst, exs, den, EE * 4);
    edge_aggr<<<EE * 32 / 256, 256, 0, stream>>>(hl, exs, den, src, dst, aggr, EE);
    residual_ln<<<NN * 32 / 256, 256, 0, stream>>>(
        h, aggr, cb + l * 128, lng + l * 128, lnb + l * 128, NN);
  }

  hipMemsetAsync(sums, 0, (size_t)GG * 128 * 4, stream);
  hipMemsetAsync(cnt,  0, (size_t)GG * 4,       stream);
  pool_sum<<<NN * 32 / 256, 256, 0, stream>>>(h, bat, sums, cnt, NN);
  value_head<<<GG * 32 / 256, 256, 0, stream>>>(sums, cnt, W1, b1, W2, b2, out, GG);
}
